// GNNGraphClass_4552665333841
// MI455X (gfx1250) — compile-verified, matrix operand order checked by reading, not run on hardware
//
#include <hip/hip_runtime.h>
#include <hip/hip_bf16.h>

typedef __attribute__((ext_vector_type(16))) _Float16 v16h;
typedef __attribute__((ext_vector_type(8)))  _Float16 v8h;
typedef __attribute__((ext_vector_type(4)))  _Float16 v4h;
typedef __attribute__((ext_vector_type(8)))  float    v8f;

#define HID 128
#define NEG_SLOPE 0.2f
#define BN_EPS 1e-5f
#define N_GRAPHS 512

// ---------------------------------------------------------------------------
// Pre-pack W[128 x N] (f32, row-major) into f16 WMMA B-fragments.
// Fragment id = (col_tile * 4 + kt) * 32 + lane ; each holds 16 halves:
//   half j   = W[kt*32      + hi*8 + j][n0+hl]
//   half 8+j = W[kt*32 + 16 + hi*8 + j][n0+hl]
// matching the ISA 7.12.2 16-bit B 32x16 layout. One thread per fragment.
// ---------------------------------------------------------------------------
__global__ void pack_w_f16(const float* __restrict__ W, _Float16* __restrict__ Wp, int N) {
    int id = blockIdx.x * blockDim.x + threadIdx.x;
    int total = (N / 16) * 4 * 32;
    if (id >= total) return;
    int t    = id >> 7;          // column tile
    int kt   = (id >> 5) & 3;    // K step (32 each)
    int lane = id & 31;
    int hl = lane & 15, hi = lane >> 4;
    int n = t * 16 + hl;
    _Float16* o = Wp + (size_t)id * 16;
#pragma unroll
    for (int j = 0; j < 8; ++j) {
        o[j]     = (_Float16)W[(size_t)(kt * 32 + hi * 8 + j) * N + n];
        o[8 + j] = (_Float16)W[(size_t)(kt * 32 + 16 + hi * 8 + j) * N + n];
    }
}

// ---------------------------------------------------------------------------
// WMMA GEMM: C[M x N] = A[M x 128] @ B[128 x N] (+ bias[N] if HAS_BIAS)
// Requires M % 16 == 0 (true for all call sites: 50000, 512).
// grid.x = M/16; blockDim.x = 32 * (N/16) waves.
// A tile staged in LDS as f16 (converted at store); B pre-packed f16 fragments.
// K loop: 4 x v_wmma_f32_16x16x32_f16 (f32 accumulate). Branch-free epilogue.
// ---------------------------------------------------------------------------
template <bool HAS_BIAS>
__global__ void gemm128_wmma(const float* __restrict__ A, const _Float16* __restrict__ Bp,
                             const float* __restrict__ bias, float* __restrict__ C,
                             int N) {
    __shared__ __align__(16) _Float16 As[16 * 128];
    const int tid  = threadIdx.x;
    const int row0 = blockIdx.x * 16;

    // cooperative stage: 512 chunks of 4 floats -> 4 halves each
    for (int c = tid; c < 512; c += blockDim.x) {
        int r   = c >> 5;            // local row 0..15
        int col = (c & 31) * 4;
        float4 f = *(const float4*)(A + (size_t)(row0 + r) * 128 + col);
        v4h h4 = { (_Float16)f.x, (_Float16)f.y, (_Float16)f.z, (_Float16)f.w };
        *(v4h*)(As + r * 128 + col) = h4;
    }
    __syncthreads();

    const int wave = tid >> 5;
    const int lane = tid & 31;
    const int hl   = lane & 15;
    const int hi   = lane >> 4;

    v8f acc = {};
#pragma unroll
    for (int kt = 0; kt < 4; ++kt) {
        // A fragment: halves [0..7] = K{kt*32+hi*8 ..}, [8..15] = K{kt*32+16+hi*8 ..}
        int base = hl * 128 + kt * 32 + hi * 8;
        v8h alo = *(const v8h*)(As + base);
        v8h ahi = *(const v8h*)(As + base + 16);
        v16h a;
#pragma unroll
        for (int j = 0; j < 8; ++j) { a[j] = alo[j]; a[8 + j] = ahi[j]; }

        // B fragment: one contiguous 32B vector load from packed weights
        v16h b = *(const v16h*)(Bp + (((size_t)wave * 4 + kt) * 32 + lane) * 16);

        acc = __builtin_amdgcn_wmma_f32_16x16x32_f16(
            false, a, false, b, (short)0, acc, false, false);
    }

    const int col = wave * 16 + hl;
    float bv = HAS_BIAS ? bias[col] : 0.0f;
#pragma unroll
    for (int r = 0; r < 8; ++r) {                    // C/D: row = r + 8*hi, col = hl
        int row = row0 + r + 8 * hi;
        C[(size_t)row * N + col] = acc[r] + bv;
    }
}

// ---------------------------------------------------------------------------
// Per-node attention scores: s_src[n] = H[n].a_src ; s_dst[n] = H[n].a_dst
// ---------------------------------------------------------------------------
__global__ void node_scores(const float* __restrict__ H, const float* __restrict__ a_src,
                            const float* __restrict__ a_dst, float* __restrict__ ss,
                            float* __restrict__ sd, int n_nodes) {
    int node = (int)((blockIdx.x * (size_t)blockDim.x + threadIdx.x) >> 5);
    int lane = threadIdx.x & 31;
    if (node >= n_nodes) return;
    const float* h = H + (size_t)node * HID;
    float s0 = 0.f, s1 = 0.f;
#pragma unroll
    for (int j = 0; j < 4; ++j) {
        int c = lane + 32 * j;
        float hv = h[c];
        s0 += hv * a_src[c];
        s1 += hv * a_dst[c];
    }
#pragma unroll
    for (int off = 16; off > 0; off >>= 1) {
        s0 += __shfl_down(s0, off, 32);
        s1 += __shfl_down(s1, off, 32);
    }
    if (lane == 0) { ss[node] = s0; sd[node] = s1; }
}

__global__ void fill_f32(float* __restrict__ p, float v, int n) {
    int i = blockIdx.x * blockDim.x + threadIdx.x;
    if (i < n) p[i] = v;
}

__device__ __forceinline__ void atomicMaxF(float* addr, float val) {
    if (val >= 0.f) atomicMax((int*)addr, __float_as_int(val));
    else            atomicMin((unsigned int*)addr, __float_as_uint(val));
}

__device__ __forceinline__ void edge_sd(int e, int n_edges,
                                        const int* __restrict__ esrc,
                                        const int* __restrict__ edst, int& s, int& d) {
    if (e < n_edges) { s = esrc[e]; d = edst[e]; }
    else             { s = e - n_edges; d = s; }       // self-loops
}

__global__ void edge_max(const int* __restrict__ esrc, const int* __restrict__ edst,
                         const float* __restrict__ ss, const float* __restrict__ sd,
                         float* __restrict__ mx, int n_edges, int etot) {
    int e = blockIdx.x * blockDim.x + threadIdx.x;
    if (e >= etot) return;
    int s, d; edge_sd(e, n_edges, esrc, edst, s, d);
    float v = ss[s] + sd[d];
    v = (v > 0.f) ? v : NEG_SLOPE * v;
    atomicMaxF(&mx[d], v);
}

__global__ void edge_expsum(const int* __restrict__ esrc, const int* __restrict__ edst,
                            const float* __restrict__ ss, const float* __restrict__ sd,
                            const float* __restrict__ mx, float* __restrict__ den,
                            int n_edges, int etot) {
    int e = blockIdx.x * blockDim.x + threadIdx.x;
    if (e >= etot) return;
    int s, d; edge_sd(e, n_edges, esrc, edst, s, d);
    float v = ss[s] + sd[d];
    v = (v > 0.f) ? v : NEG_SLOPE * v;
    atomicAdd(&den[d], __expf(v - mx[d]));
}

// One wave per edge: lane j accumulates features j, j+32, j+64, j+96 (coalesced, L2-resident)
__global__ void edge_aggregate(const float* __restrict__ H, const int* __restrict__ esrc,
                               const int* __restrict__ edst, const float* __restrict__ ss,
                               const float* __restrict__ sd, const float* __restrict__ mx,
                               const float* __restrict__ den, float* __restrict__ out,
                               int n_edges, int etot) {
    int e = (int)((blockIdx.x * (size_t)blockDim.x + threadIdx.x) >> 5);
    int lane = threadIdx.x & 31;
    if (e >= etot) return;
    int s, d; edge_sd(e, n_edges, esrc, edst, s, d);
    float v = ss[s] + sd[d];
    v = (v > 0.f) ? v : NEG_SLOPE * v;
    float w = __expf(v - mx[d]) / (den[d] + 1e-16f);
    const float* hs = H + (size_t)s * HID;
    float* od = out + (size_t)d * HID;
#pragma unroll
    for (int j = 0; j < 4; ++j) {
        int c = lane + 32 * j;
        atomicAdd(&od[c], w * hs[c]);
    }
}

__global__ void bias_bn_relu(float* __restrict__ x, const float* __restrict__ bias,
                             const float* __restrict__ gam, const float* __restrict__ bet,
                             const float* __restrict__ mean, const float* __restrict__ var,
                             int total) {
    int i = blockIdx.x * blockDim.x + threadIdx.x;
    if (i >= total) return;
    int c = i & (HID - 1);
    float v = x[i] + bias[c];
    v = (v - mean[c]) * (gam[c] * rsqrtf(var[c] + BN_EPS)) + bet[c];
    x[i] = fmaxf(v, 0.f);
}

__global__ void pool_accum(const float* __restrict__ x, const int* __restrict__ batch,
                           float* __restrict__ sums, float* __restrict__ cnt, int n_nodes) {
    int i = blockIdx.x * blockDim.x + threadIdx.x;
    if (i >= n_nodes * HID) return;
    int n = i >> 7, c = i & (HID - 1);
    int g = batch[n];
    atomicAdd(&sums[(size_t)g * HID + c], x[i]);
    if (c == 0) atomicAdd(&cnt[g], 1.0f);
}

__global__ void pool_div(float* __restrict__ sums, const float* __restrict__ cnt, int total) {
    int i = blockIdx.x * blockDim.x + threadIdx.x;
    if (i >= total) return;
    sums[i] /= fmaxf(cnt[i >> 7], 1.0f);
}

// ---------------------------------------------------------------------------
extern "C" void kernel_launch(void* const* d_in, const int* in_sizes, int n_in,
                              void* d_out, int out_size, void* d_ws, size_t ws_size,
                              hipStream_t stream) {
    const float* gx  = (const float*)d_in[0];
    const int*   ei  = (const int*)  d_in[1];
    const int*   bat = (const int*)  d_in[2];
    const float* W1  = (const float*)d_in[3];
    const float* as1 = (const float*)d_in[4];
    const float* ad1 = (const float*)d_in[5];
    const float* b1  = (const float*)d_in[6];
    const float* g1  = (const float*)d_in[7];
    const float* be1 = (const float*)d_in[8];
    const float* mu1 = (const float*)d_in[9];
    const float* va1 = (const float*)d_in[10];
    const float* W2  = (const float*)d_in[11];
    const float* as2 = (const float*)d_in[12];
    const float* ad2 = (const float*)d_in[13];
    const float* b2  = (const float*)d_in[14];
    const float* g2  = (const float*)d_in[15];
    const float* be2 = (const float*)d_in[16];
    const float* mu2 = (const float*)d_in[17];
    const float* va2 = (const float*)d_in[18];
    const float* hw  = (const float*)d_in[19];
    const float* hb  = (const float*)d_in[20];

    const int n_nodes = in_sizes[0] / HID;       // 50000 (multiple of 16)
    const int n_edges = in_sizes[1] / 2;
    const int etot    = n_edges + n_nodes;       // with self-loops
    const int* esrc = ei;
    const int* edst = ei + n_edges;

    float* ws   = (float*)d_ws;
    float* H    = ws;                                   // n_nodes*128
    float* AGG  = H    + (size_t)n_nodes * HID;         // n_nodes*128
    float* SS   = AGG  + (size_t)n_nodes * HID;
    float* SD   = SS   + n_nodes;
    float* MX   = SD   + n_nodes;
    float* DEN  = MX   + n_nodes;
    float* POOL = DEN  + n_nodes;                       // 512*128
    float* CNT  = POOL + (size_t)N_GRAPHS * HID;        // 512
    // packed-weight region, 32B aligned (round float offset up to multiple of 8)
    size_t wpOff = (size_t)(CNT + N_GRAPHS - ws);
    wpOff = (wpOff + 7) & ~(size_t)7;
    _Float16* WP = (_Float16*)(ws + wpOff);             // up to 128*128 halves (32 KB)

    const int feat_total = n_nodes * HID;
    const int gridFeat  = (feat_total + 255) / 256;
    const int gridNode  = (n_nodes + 255) / 256;
    const int gridEdge  = (etot + 255) / 256;
    const int gridEdgeW = (etot + 7) / 8;               // wave per edge
    const int gridNodeW = (n_nodes + 7) / 8;            // wave per node
    const int gridGemm  = n_nodes / 16;                 // exact

    // ---- two GAT layers -------------------------------------------------
    for (int layer = 0; layer < 2; ++layer) {
        const float* xin = (layer == 0) ? gx : AGG;     // AGG holds x1 after layer 1
        const float* W   = (layer == 0) ? W1 : W2;
        const float* asr = (layer == 0) ? as1 : as2;
        const float* ads = (layer == 0) ? ad1 : ad2;
        const float* bia = (layer == 0) ? b1  : b2;
        const float* gam = (layer == 0) ? g1  : g2;
        const float* bet = (layer == 0) ? be1 : be2;
        const float* mu  = (layer == 0) ? mu1 : mu2;
        const float* va  = (layer == 0) ? va1 : va2;

        pack_w_f16<<<4, 256, 0, stream>>>(W, WP, HID);
        gemm128_wmma<false><<<gridGemm, 256, 0, stream>>>(xin, WP, nullptr, H, HID);
        node_scores<<<gridNodeW, 256, 0, stream>>>(H, asr, ads, SS, SD, n_nodes);
        fill_f32<<<gridNode, 256, 0, stream>>>(MX, -3.402823466e38f, n_nodes);
        fill_f32<<<gridNode, 256, 0, stream>>>(DEN, 0.0f, n_nodes);
        fill_f32<<<gridFeat, 256, 0, stream>>>(AGG, 0.0f, feat_total);   // x1 consumed by gemm above
        edge_max<<<gridEdge, 256, 0, stream>>>(esrc, edst, SS, SD, MX, n_edges, etot);
        edge_expsum<<<gridEdge, 256, 0, stream>>>(esrc, edst, SS, SD, MX, DEN, n_edges, etot);
        edge_aggregate<<<gridEdgeW, 256, 0, stream>>>(H, esrc, edst, SS, SD, MX, DEN, AGG,
                                                      n_edges, etot);
        bias_bn_relu<<<gridFeat, 256, 0, stream>>>(AGG, bia, gam, bet, mu, va, feat_total);
    }

    // ---- global mean pool ----------------------------------------------
    fill_f32<<<(N_GRAPHS * HID + 255) / 256, 256, 0, stream>>>(POOL, 0.0f, N_GRAPHS * HID);
    fill_f32<<<(N_GRAPHS + 255) / 256, 256, 0, stream>>>(CNT, 0.0f, N_GRAPHS);
    pool_accum<<<gridFeat, 256, 0, stream>>>(AGG, bat, POOL, CNT, n_nodes);
    pool_div<<<(N_GRAPHS * HID + 255) / 256, 256, 0, stream>>>(POOL, CNT, N_GRAPHS * HID);

    // ---- head: [512x128] @ [128x64] + b --------------------------------
    pack_w_f16<<<2, 256, 0, stream>>>(hw, WP, 64);
    gemm128_wmma<true><<<N_GRAPHS / 16, 32 * (64 / 16), 0, stream>>>(
        POOL, WP, hb, (float*)d_out, 64);
}
